// Head_53704271069520
// MI455X (gfx1250) — matrix operand order repaired and checked by reading, not compiled, not run on hardware
//
#include <hip/hip_runtime.h>

// ---------------------------------------------------------------------------
// Fused attention head for MI455X (gfx1250, wave32, WMMA).
//   Phase 1 (proj_kernel):  q = Q@Wq, k = K@Wk, v = V@Wv   (f32 -> f16 out)
//       memory-bound: 192 MB of f32 activations read once from HBM.
//   Phase 2 (attn_kernel):  flash-attention over f16 q/k/vT (6 MB, L2-resident)
//       double-buffered GLOBAL_LOAD_ASYNC_TO_LDS staging (ASYNCcnt),
//       causal + key-pad + query-pad masking, online softmax, f32 output.
// ---------------------------------------------------------------------------

#define BATCH 8
#define SEQ   2048
#define CDIM  1024
#define HSZ   64
#define BT    (BATCH * SEQ)
#define ATT_SCALE 0.03125f   // C^-0.5 = 1/32

typedef _Float16 __attribute__((ext_vector_type(16))) v16h;
typedef _Float16 __attribute__((ext_vector_type(8)))  v8h;
typedef float    __attribute__((ext_vector_type(8)))  v8f;
typedef float    __attribute__((ext_vector_type(4)))  f4;
typedef int      __attribute__((ext_vector_type(4)))  v4i;

// neg/reuse/c_mod flags must be ICE at the call site -> macro, not function
#define WMMA_F16(a, b, c) \
    __builtin_amdgcn_wmma_f32_16x16x32_f16(false, (a), false, (b), (short)0, (c), false, false)

// ---- CDNA5 async global->LDS copy (ASYNCcnt), builtin if present else asm --
typedef __attribute__((address_space(1))) v4i* as1_v4i_p;   // global v4i*
typedef __attribute__((address_space(3))) v4i* as3_v4i_p;   // LDS v4i*

__device__ __forceinline__ void async_b128(const void* g, void* l)
{
#if __has_builtin(__builtin_amdgcn_global_load_async_to_lds_b128)
    // inttoptr conversions: AS1 value == generic value; AS3 == low 32 bits.
    __builtin_amdgcn_global_load_async_to_lds_b128(
        (as1_v4i_p)(size_t)g, (as3_v4i_p)(size_t)l, 0, 0);
#else
    const unsigned ldso = (unsigned)(size_t)l;   // LDS offset = addr[31:0]
    asm volatile("global_load_async_to_lds_b128 %0, %1, off"
                 :: "v"(ldso), "v"(g) : "memory");
#endif
}

__device__ __forceinline__ void wait_async0()
{
#if __has_builtin(__builtin_amdgcn_s_wait_asynccnt)
    __builtin_amdgcn_s_wait_asynccnt(0);
#else
    asm volatile("s_wait_asynccnt 0x0" ::: "memory");
#endif
}

// ===========================================================================
// Kernel 1: projections.  grid = (BT/128, 3), block = 256 (8 waves).
//   blockIdx.y: 0 -> q (Q@Wq), 1 -> k (K@Wk), 2 -> v (V@Wv, stored [B][HS][T])
// ===========================================================================
__global__ __launch_bounds__(256) void proj_kernel(
    const float* __restrict__ Q, const float* __restrict__ K,
    const float* __restrict__ V,
    const float* __restrict__ Wq, const float* __restrict__ Wk,
    const float* __restrict__ Wv,
    _Float16* __restrict__ qh, _Float16* __restrict__ kh,
    _Float16* __restrict__ vt)
{
    __shared__ _Float16 WT[HSZ][32];          // weight block, transposed (4 KB)

    const int tid  = threadIdx.x;
    const int wave = tid >> 5;
    const int lane = tid & 31;
    const int n16  = lane & 15;
    const int hi   = lane >> 4;
    const int kb8  = hi * 8;                  // A-fragment K base per lane half
    const int p    = blockIdx.y;

    const float* X = (p == 0) ? Q  : (p == 1) ? K  : V;
    const float* W = (p == 0) ? Wq : (p == 1) ? Wk : Wv;

    const int row0 = blockIdx.x * 128 + wave * 16;   // wave's first output row
    const int arow = row0 + n16;                     // this lane's A row

    v8f acc[4];
    #pragma unroll
    for (int nt = 0; nt < 4; ++nt)
        #pragma unroll
        for (int e = 0; e < 8; ++e) acc[nt][e] = 0.0f;

    for (int kb = 0; kb < CDIM / 32; ++kb) {
        // ---- cooperative stage: W[kb*32 .. +31][0..63] -> WT[h][j] (f16) ----
        {
            const int j  = tid >> 3;           // 0..31 source row in block
            const int h0 = (tid & 7) * 8;      // 8 consecutive columns
            const float* src = W + (size_t)(kb * 32 + j) * HSZ + h0;
            f4 w0 = *(const f4*)(src);
            f4 w1 = *(const f4*)(src + 4);
            #pragma unroll
            for (int e = 0; e < 4; ++e) WT[h0 + e][j]     = (_Float16)w0[e];
            #pragma unroll
            for (int e = 0; e < 4; ++e) WT[h0 + 4 + e][j] = (_Float16)w1[e];
        }
        __syncthreads();

        // ---- A fragment: X[arow][kb*32 + ISA 16-bit A pattern] ----
        const float* xr = X + (size_t)arow * CDIM + kb * 32;
        f4 a0 = *(const f4*)(xr + kb8);
        f4 a1 = *(const f4*)(xr + kb8 + 4);
        f4 a2 = *(const f4*)(xr + 16 + kb8);
        f4 a3 = *(const f4*)(xr + 16 + kb8 + 4);
        if (kb + 1 < CDIM / 32)
            __builtin_prefetch(xr + 32, 0, 1);           // global_prefetch_b8
        v16h afrag;
        #pragma unroll
        for (int e = 0; e < 4; ++e) {
            afrag[e]      = (_Float16)a0[e];
            afrag[4 + e]  = (_Float16)a1[e];
            afrag[8 + e]  = (_Float16)a2[e];
            afrag[12 + e] = (_Float16)a3[e];
        }

        // ---- 4 N-tiles of 16: B fragment = WT[n][kb8 + i], WT[n][16 + kb8 + i] ----
        #pragma unroll
        for (int nt = 0; nt < 4; ++nt) {
            const int n = nt * 16 + n16;
            v8h b0 = *(const v8h*)&WT[n][kb8];
            v8h b1 = *(const v8h*)&WT[n][16 + kb8];
            v16h bfrag;
            #pragma unroll
            for (int e = 0; e < 8; ++e) { bfrag[e] = b0[e]; bfrag[8 + e] = b1[e]; }
            acc[nt] = WMMA_F16(afrag, bfrag, acc[nt]);
        }
        __syncthreads();
    }

    // ---- store (f16) ----
    if (p < 2) {
        _Float16* dst = (p == 0) ? qh : kh;        // [B*T][HS] row-major
        #pragma unroll
        for (int nt = 0; nt < 4; ++nt) {
            const int h = nt * 16 + n16;
            #pragma unroll
            for (int r = 0; r < 8; ++r) {
                const int row = row0 + r + 8 * hi;
                dst[(size_t)row * HSZ + h] = (_Float16)acc[nt][r];
            }
        }
    } else {                                        // v transposed: [B][HS][T]
        const int b  = row0 / SEQ;                  // 128-row tile never splits b
        const int t0 = (row0 % SEQ) + 8 * hi;
        #pragma unroll
        for (int nt = 0; nt < 4; ++nt) {
            const int h = nt * 16 + n16;
            v8h pk;
            #pragma unroll
            for (int r = 0; r < 8; ++r) pk[r] = (_Float16)acc[nt][r];
            *(v8h*)(vt + ((size_t)b * HSZ + h) * SEQ + t0) = pk;
        }
    }
}

// ===========================================================================
// Kernel 2: flash attention.  grid = (SEQ/64, B), block = 128 (4 waves).
// Double-buffered async (global->LDS) staging of k / vT key blocks.
// ===========================================================================
__global__ __launch_bounds__(128) void attn_kernel(
    const _Float16* __restrict__ qh, const _Float16* __restrict__ kh,
    const _Float16* __restrict__ vt, const unsigned char* __restrict__ pad,
    float* __restrict__ out)
{
    __shared__ _Float16 kbuf[2][32][HSZ];     // key blocks   [j][d]   (8 KB)
    __shared__ _Float16 vbuf[2][HSZ][32];     // value blocks [h][j]   (8 KB)
    __shared__ _Float16 pbuf[4][16][32];      // per-wave P tile       (4 KB)

    const int tid   = threadIdx.x;
    const int wave  = tid >> 5;
    const int lane  = tid & 31;
    const int n16   = lane & 15;
    const int hi    = lane >> 4;
    const int kb8   = hi * 8;
    const int b     = blockIdx.y;
    const int qbase = blockIdx.x * 64;        // workgroup's first query row
    const int qrow0 = qbase + wave * 16;      // wave's first query row

    // ---- q A-fragments for both 32-wide d blocks (held in VGPRs) ----
    const _Float16* qr = qh + ((size_t)b * SEQ + qrow0 + n16) * HSZ;
    v16h qfrag[2];
    #pragma unroll
    for (int db = 0; db < 2; ++db) {
        v8h c0 = *(const v8h*)(qr + db * 32 + kb8);
        v8h c1 = *(const v8h*)(qr + db * 32 + 16 + kb8);
        #pragma unroll
        for (int e = 0; e < 8; ++e) { qfrag[db][e] = c0[e]; qfrag[db][8 + e] = c1[e]; }
    }

    bool qpad[8];
    #pragma unroll
    for (int r = 0; r < 8; ++r)
        qpad[r] = pad[(size_t)b * SEQ + qrow0 + r + 8 * hi] != 0;

    v8f oacc[4];
    #pragma unroll
    for (int h = 0; h < 4; ++h)
        #pragma unroll
        for (int e = 0; e < 8; ++e) oacc[h][e] = 0.0f;

    float mrow[8], lrow[8];
    #pragma unroll
    for (int r = 0; r < 8; ++r) { mrow[r] = -1e30f; lrow[r] = 0.0f; }

    const float NEG_INF = -__builtin_inff();
    const int nkb = (qbase + 64) / 32;        // causal key-block limit

    // per-thread async staging of one 32-key block into LDS buffer `bsel`
    auto stage = [&](int kb, int bsel) {
        const int j0 = kb * 32;
        {   // k block: 32 rows x 128 B
            const int row = tid >> 2;
            const int c0  = (tid & 3) * 16;
            const _Float16* src = kh + ((size_t)b * SEQ + j0 + row) * HSZ + c0;
            async_b128(src,     &kbuf[bsel][row][c0]);
            async_b128(src + 8, &kbuf[bsel][row][c0 + 8]);
        }
        {   // vT block: 64 rows x 64 B
            const int h  = tid >> 1;
            const int c0 = (tid & 1) * 16;
            const _Float16* src = vt + ((size_t)b * HSZ + h) * SEQ + j0 + c0;
            async_b128(src,     &vbuf[bsel][h][c0]);
            async_b128(src + 8, &vbuf[bsel][h][c0 + 8]);
        }
    };

    stage(0, 0);             // prologue fill
    wait_async0();
    __syncthreads();

    for (int kb = 0; kb < nkb; ++kb) {
        const int cur = kb & 1;
        const int j0  = kb * 32;

        if (kb + 1 < nkb)                 // overlap next fill with this compute
            stage(kb + 1, cur ^ 1);

        // ---- S = q @ k^T : two 16x16 f32 tiles (keys j0..+15, j0+16..+31) ----
        float sv[2][8];
        #pragma unroll
        for (int t = 0; t < 2; ++t) {
            v8f s;
            #pragma unroll
            for (int e = 0; e < 8; ++e) s[e] = 0.0f;
            #pragma unroll
            for (int db = 0; db < 2; ++db) {
                v8h b0 = *(const v8h*)&kbuf[cur][t * 16 + n16][db * 32 + kb8];
                v8h b1 = *(const v8h*)&kbuf[cur][t * 16 + n16][db * 32 + 16 + kb8];
                v16h bfrag;
                #pragma unroll
                for (int e = 0; e < 8; ++e) { bfrag[e] = b0[e]; bfrag[8 + e] = b1[e]; }
                s = WMMA_F16(qfrag[db], bfrag, s);
            }
            // masking (reference order: causal -inf, key-pad -inf, then
            // query-pad rows overwritten with -1e9)
            const int  jg = j0 + t * 16 + n16;
            const bool kp = pad[(size_t)b * SEQ + jg] != 0;
            #pragma unroll
            for (int r = 0; r < 8; ++r) {
                const int ig = qrow0 + r + 8 * hi;
                float x = s[r] * ATT_SCALE;
                if (jg > ig || kp) x = NEG_INF;
                if (qpad[r])       x = -1e9f;
                sv[t][r] = x;
            }
        }

        // ---- online softmax: row reductions across the 16-lane N group ----
        float pj[2][8];
        #pragma unroll
        for (int r = 0; r < 8; ++r) {
            float bm = fmaxf(sv[0][r], sv[1][r]);
            bm = fmaxf(bm, __shfl_xor(bm, 1));
            bm = fmaxf(bm, __shfl_xor(bm, 2));
            bm = fmaxf(bm, __shfl_xor(bm, 4));
            bm = fmaxf(bm, __shfl_xor(bm, 8));
            const float mnew  = fmaxf(mrow[r], bm);
            const float alpha = __expf(mrow[r] - mnew);
            mrow[r] = mnew;
            const float p0 = __expf(sv[0][r] - mnew);
            const float p1 = __expf(sv[1][r] - mnew);
            pj[0][r] = p0;  pj[1][r] = p1;
            float ps = p0 + p1;
            ps += __shfl_xor(ps, 1);
            ps += __shfl_xor(ps, 2);
            ps += __shfl_xor(ps, 4);
            ps += __shfl_xor(ps, 8);
            lrow[r] = lrow[r] * alpha + ps;
            #pragma unroll
            for (int h = 0; h < 4; ++h) oacc[h][r] *= alpha;
        }

        // ---- C-layout -> A-layout for P via per-wave LDS tile ----
        #pragma unroll
        for (int t = 0; t < 2; ++t)
            #pragma unroll
            for (int r = 0; r < 8; ++r)
                pbuf[wave][r + 8 * hi][t * 16 + n16] = (_Float16)pj[t][r];
        __syncthreads();

        v16h pfrag;
        {
            v8h p0 = *(const v8h*)&pbuf[wave][n16][kb8];
            v8h p1 = *(const v8h*)&pbuf[wave][n16][16 + kb8];
            #pragma unroll
            for (int e = 0; e < 8; ++e) { pfrag[e] = p0[e]; pfrag[8 + e] = p1[e]; }
        }

        // ---- O += P @ v : 4 head tiles ----
        #pragma unroll
        for (int h = 0; h < 4; ++h) {
            v8h b0 = *(const v8h*)&vbuf[cur][h * 16 + n16][kb8];
            v8h b1 = *(const v8h*)&vbuf[cur][h * 16 + n16][16 + kb8];
            v16h bfrag;
            #pragma unroll
            for (int e = 0; e < 8; ++e) { bfrag[e] = b0[e]; bfrag[8 + e] = b1[e]; }
            oacc[h] = WMMA_F16(pfrag, bfrag, oacc[h]);
        }

        wait_async0();       // next block's async fill has landed in LDS
        __syncthreads();     // everyone done reading `cur` before it is reused
    }

    // ---- finalize: O / l  -> f32 output [B][T][HS] ----
    #pragma unroll
    for (int h = 0; h < 4; ++h) {
        const int hc = h * 16 + n16;
        #pragma unroll
        for (int r = 0; r < 8; ++r) {
            const int ig = qrow0 + r + 8 * hi;
            out[((size_t)b * SEQ + ig) * HSZ + hc] = oacc[h][r] / lrow[r];
        }
    }
}

// ===========================================================================
extern "C" void kernel_launch(void* const* d_in, const int* in_sizes, int n_in,
                              void* d_out, int out_size, void* d_ws, size_t ws_size,
                              hipStream_t stream)
{
    (void)in_sizes; (void)n_in; (void)out_size; (void)ws_size;

    const float* V  = (const float*)d_in[0];
    const float* K  = (const float*)d_in[1];
    const float* Q  = (const float*)d_in[2];
    const float* Wq = (const float*)d_in[3];
    const float* Wk = (const float*)d_in[4];
    const float* Wv = (const float*)d_in[5];
    const unsigned char* pad = (const unsigned char*)d_in[6];   // [B,1,T] bool

    _Float16* ws = (_Float16*)d_ws;                 // 6 MB of f16 scratch
    _Float16* qh = ws;                              // [B*T][HS]
    _Float16* kh = ws + (size_t)BT * HSZ;           // [B*T][HS]
    _Float16* vt = ws + (size_t)2 * BT * HSZ;       // [B][HS][T]

    proj_kernel<<<dim3(BT / 128, 3), 256, 0, stream>>>(Q, K, V, Wq, Wk, Wv,
                                                       qh, kh, vt);
    attn_kernel<<<dim3(SEQ / 64, BATCH), 128, 0, stream>>>(qh, kh, vt, pad,
                                                           (float*)d_out);
}
